// MogrifierLSTMLayer_70824010711232
// MI455X (gfx1250) — matrix operand order repaired and checked by reading, not compiled, not run on hardware
//
#include <hip/hip_runtime.h>
#include <hip/hip_bf16.h>
#include <stdint.h>

// ---------------- problem constants ----------------
constexpr int HDIM  = 1024;
constexpr int BATCH = 8192;
constexpr int FDIM  = 4 * HDIM;   // 4096
constexpr float LN_EPS = 1e-6f;

// ---------------- WMMA types ----------------
typedef __attribute__((ext_vector_type(16))) __bf16 v16bf;
typedef __attribute__((ext_vector_type(8)))  float  v8f;

#define AS1 __attribute__((address_space(1)))
#define AS3 __attribute__((address_space(3)))

// Matches the builtin's parameter type: 'int __vector(4)' pointer
typedef int v4i_t __attribute__((vector_size(16)));

// Async global->LDS copy path (CDNA5): guarded so the file always compiles.
#if defined(__has_builtin)
#  if __has_builtin(__builtin_amdgcn_global_load_async_to_lds_b128)
#    define HAVE_ASYNC_COPY 1
#  endif
#endif

__device__ __forceinline__ void wait_async0() {
#if defined(__has_builtin)
#  if __has_builtin(__builtin_amdgcn_s_wait_asynccnt)
  __builtin_amdgcn_s_wait_asynccnt(0);
  return;
#  else
  asm volatile("s_wait_asynccnt 0x0" ::: "memory");
  return;
#  endif
#else
  asm volatile("s_wait_asynccnt 0x0" ::: "memory");
#endif
}

// ---------------- helpers ----------------
__device__ __forceinline__ unsigned short f2bf(float f) {
  union { float f; uint32_t u; } c; c.f = f;
  uint32_t u = c.u;
  uint32_t r = (u + 0x7FFFu + ((u >> 16) & 1u)) >> 16;  // round-to-nearest-even
  return (unsigned short)r;
}
__device__ __forceinline__ float sigmoidf_(float x) { return 1.0f / (1.0f + __expf(-x)); }

// ---------------- tiled bf16 WMMA GEMM ----------------
// C[M,N] = epilogue( A[M,K](bf16) @ B[K,N](bf16) )
// Requirements (guaranteed by caller): M%128==0, N%256==0, K%32==0.
constexpr int BM = 128, BN = 256, BK = 32;
constexpr int LDA_S = BK + 8;   // 40 ushorts  (80 B rows: 16B-aligned, spread banks)
constexpr int LDB_S = BN + 8;   // 264 ushorts (528 B rows: 16B-aligned)

enum { EPI_STORE = 0, EPI_ACC = 1, EPI_MOG = 2, EPI_RELU_BF = 3, EPI_ADD2 = 4 };

template <int EPI>
__global__ __launch_bounds__(256) void gemm_bf16(
    const unsigned short* __restrict__ A, const unsigned short* __restrict__ Bw,
    int M, int N, int K,
    const float* __restrict__ bias,
    float* __restrict__ Cout, const float* __restrict__ Cin,
    float* __restrict__ tgtF, unsigned short* __restrict__ tgtBF,
    const float* __restrict__ add1, const float* __restrict__ add2)
{
  __shared__ unsigned short As[2][BM * LDA_S];   // 2 x 10240 B
  __shared__ unsigned short Bs[2][BK * LDB_S];   // 2 x 16896 B

  const int tid  = threadIdx.x;
  const int lane = tid & 31;
  const int wave = tid >> 5;
  const int wm   = wave & 1;   // 2 wave-rows of 64
  const int wn   = wave >> 1;  // 4 wave-cols of 64
  const int bm0  = blockIdx.y * BM;
  const int bn0  = blockIdx.x * BN;

  // global->LDS staging maps (uint4 = 8 bf16)
  const int arow = tid >> 2;        // 0..63, two sweeps of 64 rows cover 128
  const int acol = (tid & 3) * 8;   // covers 32 cols
  const int brow = tid >> 5;        // 0..7, four sweeps of 8 rows cover 32
  const int bcol = (tid & 31) * 8;  // covers 256 cols

  // pointer-increment addressing: one add per k-tile, no per-iteration muls
  const unsigned short* gA = A  + (size_t)(bm0 + arow) * K + acol;
  const unsigned short* gB = Bw + (size_t)brow * N + bn0 + bcol;
  const size_t gAstep = (size_t)64 * K;  // A sweep stride (rows)
  const size_t gBstep = (size_t)8 * N;   // B sweep stride (rows)

  auto loadTile = [&](int buf) {
#if defined(HAVE_ASYNC_COPY)
#pragma unroll
    for (int s = 0; s < 2; ++s) {
      __builtin_amdgcn_global_load_async_to_lds_b128(
          (AS1 v4i_t*)(gA + s * gAstep),
          (AS3 v4i_t*)&As[buf][(arow + s * 64) * LDA_S + acol], 0, 0);
    }
#pragma unroll
    for (int s = 0; s < 4; ++s) {
      __builtin_amdgcn_global_load_async_to_lds_b128(
          (AS1 v4i_t*)(gB + s * gBstep),
          (AS3 v4i_t*)&Bs[buf][(brow + s * 8) * LDB_S + bcol], 0, 0);
    }
#else
#pragma unroll
    for (int s = 0; s < 2; ++s) {
      uint4 q = *reinterpret_cast<const uint4*>(gA + s * gAstep);
      *reinterpret_cast<uint4*>(&As[buf][(arow + s * 64) * LDA_S + acol]) = q;
    }
#pragma unroll
    for (int s = 0; s < 4; ++s) {
      uint4 q = *reinterpret_cast<const uint4*>(gB + s * gBstep);
      *reinterpret_cast<uint4*>(&Bs[buf][(brow + s * 8) * LDB_S + bcol]) = q;
    }
#endif
    gA += BK;
    gB += (size_t)BK * N;
  };

  auto ldfrag = [&](const unsigned short* p0, const unsigned short* p1) -> v16bf {
    union { uint4 q[2]; v16bf v; } u;
    u.q[0] = *reinterpret_cast<const uint4*>(p0);
    u.q[1] = *reinterpret_cast<const uint4*>(p1);
    return u.v;
  };

  v8f acc[4][4];
#pragma unroll
  for (int mi = 0; mi < 4; ++mi)
#pragma unroll
    for (int ni = 0; ni < 4; ++ni) {
      v8f z = {0.f, 0.f, 0.f, 0.f, 0.f, 0.f, 0.f, 0.f};
      acc[mi][ni] = z;
    }

  const int nK = K / BK;
  loadTile(0);
#if defined(HAVE_ASYNC_COPY)
  wait_async0();
#endif
  __syncthreads();

  for (int kt = 0; kt < nK; ++kt) {
    const int cur = kt & 1;
    if (kt + 1 < nK) loadTile(cur ^ 1);
    if (kt + 2 < nK) {  // L2 prefetch two tiles ahead -> global_prefetch_b8
      __builtin_prefetch(gA, 0, 1);
      __builtin_prefetch(gB, 0, 1);
    }

    const int khalf = lane >> 4;     // ISA A-layout: lanes 0-15 K=0-7/16-23, 16-31 K=8-15/24-31
    const int l15   = lane & 15;

    v16bf afr[4];
#pragma unroll
    for (int mi = 0; mi < 4; ++mi) {
      const unsigned short* base = &As[cur][(wm * 64 + mi * 16 + l15) * LDA_S + khalf * 8];
      afr[mi] = ldfrag(base, base + 16);
    }
    v16bf bfr[4];
#pragma unroll
    for (int ni = 0; ni < 4; ++ni) {
      const unsigned short* base = &Bs[cur][(khalf * 8) * LDB_S + wn * 64 + ni * 16 + l15];
      union { unsigned short e[16]; v16bf v; } u;
#pragma unroll
      for (int s = 0; s < 8; ++s) {
        u.e[s]     = base[s * LDB_S];
        u.e[s + 8] = base[(16 + s) * LDB_S];
      }
      bfr[ni] = u.v;
    }

#pragma unroll
    for (int mi = 0; mi < 4; ++mi)
#pragma unroll
      for (int ni = 0; ni < 4; ++ni)
        acc[mi][ni] = __builtin_amdgcn_wmma_f32_16x16x32_bf16(
            false, afr[mi], false, bfr[ni], (short)0, acc[mi][ni], false, false);

#if defined(HAVE_ASYNC_COPY)
    wait_async0();   // next tile resident in LDS before the barrier releases waves
#endif
    __syncthreads();
  }

  // epilogue: C/D layout -> VGPR r: M = r + 8*(lane>=16), N = lane&15
  const int mrow = (lane >> 4) << 3;
  const int ncol = lane & 15;
#pragma unroll
  for (int mi = 0; mi < 4; ++mi) {
#pragma unroll
    for (int ni = 0; ni < 4; ++ni) {
#pragma unroll
      for (int r = 0; r < 8; ++r) {
        const int gm = bm0 + wm * 64 + mi * 16 + r + mrow;
        const int gn = bn0 + wn * 64 + ni * 16 + ncol;
        const size_t idx = (size_t)gm * N + gn;
        float v = acc[mi][ni][r];
        if (EPI == EPI_STORE) {
          if (bias) v += bias[gn];
          Cout[idx] = v;
        } else if (EPI == EPI_ACC) {
          Cout[idx] = v + Cin[idx];
        } else if (EPI == EPI_MOG) {
          const float g = 2.0f * sigmoidf_(v + bias[gn]);
          const float o = tgtF[idx] * g;
          tgtF[idx]  = o;
          tgtBF[idx] = f2bf(o);
        } else if (EPI == EPI_RELU_BF) {
          float rv = v + bias[gn];
          rv = rv > 0.f ? rv : 0.f;
          tgtBF[idx] = f2bf(rv);
        } else {  // EPI_ADD2
          Cout[idx] = v + bias[gn] + add1[idx] + add2[idx];
        }
      }
    }
  }
}

// ---------------- elementwise kernels ----------------
__global__ void cvt_bf16_kernel(const float* __restrict__ in, unsigned short* __restrict__ out, int n) {
  const int i = blockIdx.x * 256 + threadIdx.x;
  if (i < n) out[i] = f2bf(in[i]);
}

__global__ void init_h_kernel(const float* __restrict__ hp, float* __restrict__ hm,
                              unsigned short* __restrict__ hbf, int n) {
  const int i = blockIdx.x * 256 + threadIdx.x;
  if (i < n) { const float v = hp[i]; hm[i] = v; hbf[i] = f2bf(v); }
}

__global__ __launch_bounds__(256) void layernorm_kernel(
    const float* __restrict__ x, const float* __restrict__ gamma, const float* __restrict__ beta,
    float* __restrict__ xn, float* __restrict__ xmog,
    unsigned short* __restrict__ xnbf, unsigned short* __restrict__ xbf)
{
  const int row = blockIdx.x, tid = threadIdx.x;
  const float* xr = x + (size_t)row * HDIM;
  float v[4], s = 0.f;
#pragma unroll
  for (int i = 0; i < 4; ++i) { v[i] = xr[tid + i * 256]; s += v[i]; }

  __shared__ float red[8];
  __shared__ float mu_s, inv_s;
  for (int off = 16; off > 0; off >>= 1) s += __shfl_down(s, off, 32);
  if ((tid & 31) == 0) red[tid >> 5] = s;
  __syncthreads();
  if (tid == 0) { float t = 0.f; for (int i = 0; i < 8; ++i) t += red[i]; mu_s = t / (float)HDIM; }
  __syncthreads();
  const float mu = mu_s;

  float s2 = 0.f;
#pragma unroll
  for (int i = 0; i < 4; ++i) { const float d = v[i] - mu; s2 += d * d; }
  for (int off = 16; off > 0; off >>= 1) s2 += __shfl_down(s2, off, 32);
  __syncthreads();
  if ((tid & 31) == 0) red[tid >> 5] = s2;
  __syncthreads();
  if (tid == 0) {
    float t = 0.f; for (int i = 0; i < 8; ++i) t += red[i];
    inv_s = 1.0f / (sqrtf(t / (float)HDIM) + LN_EPS);
  }
  __syncthreads();
  const float inv = inv_s;

#pragma unroll
  for (int i = 0; i < 4; ++i) {
    const int c = tid + i * 256;
    const float y = gamma[c] * ((v[i] - mu) * inv) + beta[c];
    const size_t idx = (size_t)row * HDIM + c;
    xn[idx] = y; xmog[idx] = y; xnbf[idx] = f2bf(y); xbf[idx] = f2bf(y);
  }
}

__global__ void gates_kernel(const float* __restrict__ lin, const float* __restrict__ Wb,
                             const float* __restrict__ Ub, const float* __restrict__ bb,
                             const float* __restrict__ cprev,
                             float* __restrict__ cnext, float* __restrict__ hnext,
                             unsigned short* __restrict__ hbf)
{
  const int i = blockIdx.x * 256 + threadIdx.x;   // grid sized exactly to BATCH*HDIM
  const int b = i / HDIM, j = i % HDIM;
  const size_t base = (size_t)b * FDIM;
  const float li = lin[base + j]            + Wb[j]            + Ub[j]            + bb[j];
  const float lf = lin[base + HDIM + j]     + Wb[HDIM + j]     + Ub[HDIM + j]     + bb[HDIM + j];
  const float lo = lin[base + 2 * HDIM + j] + Wb[2 * HDIM + j] + Ub[2 * HDIM + j] + bb[2 * HDIM + j];
  const float lg = lin[base + 3 * HDIM + j] + Wb[3 * HDIM + j] + Ub[3 * HDIM + j] + bb[3 * HDIM + j];
  const float gi = sigmoidf_(li), gf = sigmoidf_(lf), go = sigmoidf_(lo);
  const float c = gf * cprev[i] + gi * tanhf(lg);
  const float h = go * tanhf(c);
  cnext[i] = c; hnext[i] = h; hbf[i] = f2bf(h);
}

// ---------------- host orchestration ----------------
extern "C" void kernel_launch(void* const* d_in, const int* in_sizes, int n_in,
                              void* d_out, int out_size, void* d_ws, size_t ws_size,
                              hipStream_t stream)
{
  const float* x_curr = (const float*)d_in[0];
  const float* c_prev = (const float*)d_in[1];
  const float* h_prev = (const float*)d_in[2];
  const float* ln_g   = (const float*)d_in[3];
  const float* ln_b   = (const float*)d_in[4];
  const float* Wm     = (const float*)d_in[5];
  const float* bm     = (const float*)d_in[6];
  const float* Ww     = (const float*)d_in[7];
  const float* Wb     = (const float*)d_in[8];
  const float* Uw     = (const float*)d_in[9];
  const float* Ub     = (const float*)d_in[10];
  const float* bb     = (const float*)d_in[11];
  const float* f1w    = (const float*)d_in[12];
  const float* f1b    = (const float*)d_in[13];
  const float* f2w    = (const float*)d_in[14];
  const float* f2b    = (const float*)d_in[15];
  const float* ow     = (const float*)d_in[16];
  const float* ob     = (const float*)d_in[17];

  const size_t BH = (size_t)BATCH * HDIM;
  float* out_c = (float*)d_out;
  float* out_h = out_c + BH;
  float* out_y = out_c + 2 * BH;

  char* ws = (char*)d_ws;
  size_t off = 0;
  auto carve = [&](size_t bytes) -> void* {
    void* p = ws + off;
    off += (bytes + 255) & ~(size_t)255;
    return p;
  };
  float* x_norm = (float*)carve(BH * 4);
  float* x_mog  = (float*)carve(BH * 4);
  float* h_mog  = (float*)carve(BH * 4);
  float* y_ffwd = (float*)carve(BH * 4);
  float* lin    = (float*)carve((size_t)BATCH * FDIM * 4);
  unsigned short* xn_bf = (unsigned short*)carve(BH * 2);
  unsigned short* x_bf  = (unsigned short*)carve(BH * 2);
  unsigned short* h_bf  = (unsigned short*)carve(BH * 2);
  unsigned short* hn_bf = (unsigned short*)carve(BH * 2);
  unsigned short* a1_bf = (unsigned short*)carve((size_t)BATCH * FDIM * 2);
  unsigned short* WmB   = (unsigned short*)carve(5ull * HDIM * HDIM * 2);
  unsigned short* WwB   = (unsigned short*)carve((size_t)HDIM * FDIM * 2);
  unsigned short* UwB   = (unsigned short*)carve((size_t)HDIM * FDIM * 2);
  unsigned short* f1wB  = (unsigned short*)carve((size_t)HDIM * FDIM * 2);
  unsigned short* f2wB  = (unsigned short*)carve((size_t)FDIM * HDIM * 2);
  unsigned short* owB   = (unsigned short*)carve((size_t)HDIM * HDIM * 2);

  auto cvt = [&](const float* src, unsigned short* dst, size_t n) {
    cvt_bf16_kernel<<<(unsigned)((n + 255) / 256), 256, 0, stream>>>(src, dst, (int)n);
  };
  cvt(Wm,  WmB,  5ull * HDIM * HDIM);
  cvt(Ww,  WwB,  (size_t)HDIM * FDIM);
  cvt(Uw,  UwB,  (size_t)HDIM * FDIM);
  cvt(f1w, f1wB, (size_t)HDIM * FDIM);
  cvt(f2w, f2wB, (size_t)FDIM * HDIM);
  cvt(ow,  owB,  (size_t)HDIM * HDIM);

  layernorm_kernel<<<BATCH, 256, 0, stream>>>(x_curr, ln_g, ln_b, x_norm, x_mog, xn_bf, x_bf);
  init_h_kernel<<<(unsigned)(BH / 256), 256, 0, stream>>>(h_prev, h_mog, h_bf, (int)BH);

  const dim3 gH(HDIM / BN, BATCH / BM);   // (4, 64)
  const dim3 gF(FDIM / BN, BATCH / BM);   // (16, 64)

  // Mogrifier rounds (gating fused into GEMM epilogue)
  for (int r = 0; r < 5; ++r) {
    const unsigned short* Abuf = (r % 2 == 0) ? x_bf : h_bf;
    float*          tF = (r % 2 == 0) ? h_mog : x_mog;
    unsigned short* tB = (r % 2 == 0) ? h_bf : x_bf;
    gemm_bf16<EPI_MOG><<<gH, 256, 0, stream>>>(
        Abuf, WmB + (size_t)r * HDIM * HDIM, BATCH, HDIM, HDIM,
        bm + (size_t)r * HDIM, nullptr, nullptr, tF, tB, nullptr, nullptr);
  }

  // LSTM gate projections: lin = x_mog@Ww + h_mog@Uw  (biases added in gates kernel)
  gemm_bf16<EPI_STORE><<<gF, 256, 0, stream>>>(
      x_bf, WwB, BATCH, FDIM, HDIM, nullptr, lin, nullptr, nullptr, nullptr, nullptr, nullptr);
  gemm_bf16<EPI_ACC><<<gF, 256, 0, stream>>>(
      h_bf, UwB, BATCH, FDIM, HDIM, nullptr, lin, lin, nullptr, nullptr, nullptr, nullptr);

  gates_kernel<<<(unsigned)(BH / 256), 256, 0, stream>>>(
      lin, Wb, Ub, bb, c_prev, out_c, out_h, hn_bf);

  // Feedforward: a1 = relu(x_norm@f1w + f1b) ; y_ffwd = a1@f2w + f2b
  gemm_bf16<EPI_RELU_BF><<<gF, 256, 0, stream>>>(
      xn_bf, f1wB, BATCH, FDIM, HDIM, f1b, nullptr, nullptr, nullptr, a1_bf, nullptr, nullptr);
  gemm_bf16<EPI_STORE><<<gH, 256, 0, stream>>>(
      a1_bf, f2wB, BATCH, HDIM, FDIM, f2b, y_ffwd, nullptr, nullptr, nullptr, nullptr, nullptr);

  // Output projection + residual sum: y = h_next@ow + ob + x_mog + y_ffwd
  gemm_bf16<EPI_ADD2><<<gH, 256, 0, stream>>>(
      hn_bf, owB, BATCH, HDIM, HDIM, ob, out_y, nullptr, nullptr, nullptr, x_mog, y_ffwd);
}
